// GraphQNetwork_8607114461244
// MI455X (gfx1250) — matrix operand order repaired and checked by reading, not compiled, not run on hardware
//
#include <hip/hip_runtime.h>
#include <cstdint>

// ---------------- problem constants ----------------
constexpr int N_UE = 10000, N_GNB = 100, E = 320000;
constexpr int UE_F = 64, GNB_F = 32, H = 128, HEADS = 2, EDGE_F = 3, N_ACT = 32;
constexpr int NN = N_UE + N_GNB;          // 10100
constexpr int NPAD = 10112;               // next multiple of 16
constexpr int GNB_PAD = 112;              // pad 100 -> 112 rows
constexpr float NEG_SLOPE = 0.2f;

typedef __attribute__((ext_vector_type(16))) __bf16 v16bf;
typedef __attribute__((ext_vector_type(8)))  __bf16 v8bf;
typedef __attribute__((ext_vector_type(8)))  float  v8f;

#define CDIV(a, b) (((a) + (b) - 1) / (b))
static constexpr int BLK = 256;

// ---- monotonic float<->uint mapping for atomicMax on floats ----
__device__ __forceinline__ unsigned fenc(float x) {
  unsigned u = __float_as_uint(x);
  return (u & 0x80000000u) ? ~u : (u | 0x80000000u);
}
__device__ __forceinline__ float fdec(unsigned k) {
  return __uint_as_float((k & 0x80000000u) ? (k & 0x7fffffffu) : ~k);
}

// =====================================================================
// WMMA GEMM:  C[M,N] = act( A[M,K](bf16,row-major) * Bcm[N,K](bf16) + bias )
// Bcm = weights stored column-major (contiguous K per output column).
// Each wave computes a 16 x (16*NT) tile: one A fragment is reused by NT
// accumulators, stepping K by 32 with v_wmma_f32_16x16x32_bf16.
// K and NT are compile-time -> fully unrolled, branch-free WMMA chain.
// =====================================================================
template <int K, int NT>
__global__ void k_wmma_gemm(const __bf16* __restrict__ A, int lda,
                            const __bf16* __restrict__ Bcm,
                            const float* __restrict__ bias,
                            float* __restrict__ Cf, __bf16* __restrict__ Cb,
                            int ldc, int Mtiles, int NtGroups, int Mstore,
                            int doRelu) {
  int wid = (blockIdx.x * blockDim.x + threadIdx.x) >> 5;
  if (wid >= Mtiles * NtGroups) return;        // wave-uniform exit
  int lane = threadIdx.x & 31;
  int half = lane >> 4;                        // K-half selector
  int l15  = lane & 15;
  int mt = wid / NtGroups, ng = wid % NtGroups;

  // A: lane l15 owns row M; element e -> K = k0+16*half+e
  //                         (16 contiguous K starting at k0+16*half)
  const __bf16* arow = A + (size_t)(mt * 16 + l15) * lda + 16 * half;
  // B: lane l15 owns column N; 16 contiguous K starting at k0+16*half
  const __bf16* bbase = Bcm + (size_t)(ng * NT * 16 + l15) * K + 16 * half;

  v8f acc[NT];
#pragma unroll
  for (int j = 0; j < NT; ++j)
#pragma unroll
    for (int i = 0; i < 8; ++i) acc[j][i] = 0.0f;

#pragma unroll
  for (int k0 = 0; k0 < K; k0 += 32) {
    v8bf alo = *(const v8bf*)(arow + k0);
    v8bf ahi = *(const v8bf*)(arow + k0 + 8);
    v16bf av;
#pragma unroll
    for (int i = 0; i < 8; ++i) { av[i] = alo[i]; av[i + 8] = ahi[i]; }
#pragma unroll
    for (int j = 0; j < NT; ++j) {
      v16bf bv = *(const v16bf*)(bbase + (size_t)j * 16 * K + k0);
      acc[j] = __builtin_amdgcn_wmma_f32_16x16x32_bf16(
          /*neg_a=*/false, av, /*neg_b=*/false, bv,
          /*c_mod=*/(short)0, acc[j], /*reuse_a=*/false, /*reuse_b=*/false);
    }
  }

#pragma unroll
  for (int j = 0; j < NT; ++j) {
    int col = (ng * NT + j) * 16 + l15;
    float bz = bias ? bias[col] : 0.0f;
#pragma unroll
    for (int r = 0; r < 8; ++r) {
      int row = mt * 16 + r + 8 * half;   // C layout: VGPR r -> M=r / M=8+r
      if (row < Mstore) {
        float v = acc[j][r] + bz;
        if (doRelu) v = fmaxf(v, 0.0f);
        if (Cf) Cf[(size_t)row * ldc + col] = v;
        if (Cb) Cb[(size_t)row * ldc + col] = (__bf16)v;
      }
    }
  }
}

template <int K, int NT>
static void launch_gemm(const __bf16* A, int lda, const __bf16* B,
                        const float* bias, float* Cf, __bf16* Cb, int N, int M,
                        int Mstore, int relu, hipStream_t stream) {
  int mt = M / 16, ng = N / (16 * NT);
  int waves = mt * ng;
  k_wmma_gemm<K, NT><<<CDIV(waves * 32, BLK), BLK, 0, stream>>>(
      A, lda, B, bias, Cf, Cb, N, mt, ng, Mstore, relu);
}

// ---- normalize node features -> bf16 staging (zero pad rows) ----
__global__ void k_stage_rows(const float* __restrict__ X,
                             const float* __restrict__ scale,
                             const float* __restrict__ shift,
                             __bf16* __restrict__ out, int rows, int rowsPad,
                             int F) {
  int idx = blockIdx.x * blockDim.x + threadIdx.x;
  if (idx >= rowsPad * F) return;
  int r = idx / F, c = idx % F;
  float v = 0.0f;
  if (r < rows)
    v = (X[(size_t)r * F + c] - shift[c]) / (fabsf(scale[c]) + 1e-8f);
  out[idx] = (__bf16)v;
}

// ---- normalize edge attributes ----
__global__ void k_edge_norm(const float* __restrict__ ea,
                            const float* __restrict__ scale,
                            const float* __restrict__ shift,
                            float* __restrict__ out) {
  int idx = blockIdx.x * blockDim.x + threadIdx.x;
  if (idx >= E * EDGE_F) return;
  int c = idx % EDGE_F;
  out[idx] = (ea[idx] - shift[c]) / (fabsf(scale[c]) + 1e-8f);
}

// ---- convert a weight matrix (K x N, row-major f32) to column-major bf16 ----
__global__ void k_convert_w(const float* __restrict__ W, __bf16* __restrict__ o,
                            int K, int N) {
  int idx = blockIdx.x * blockDim.x + threadIdx.x;
  if (idx >= K * N) return;
  int n = idx / K, k = idx % K;
  o[idx] = (__bf16)W[(size_t)k * N + n];
}

// ---- zero pad rows of x_bf16 (rows NN..NPAD-1) ----
__global__ void k_zero_pad(__bf16* __restrict__ xb) {
  int idx = blockIdx.x * blockDim.x + threadIdx.x;
  if (idx >= (NPAD - NN) * H) return;
  xb[(size_t)NN * H + idx] = (__bf16)0.0f;
}

// ---- vhead[hd*3+j] = sum_c W_edge[j, hd*H+c] * att[hd, 2H+c] ----
__global__ void k_vhead(const float* __restrict__ Wedge,
                        const float* __restrict__ att,
                        float* __restrict__ vh) {
  int j = threadIdx.x;
  if (j >= HEADS * EDGE_F) return;
  int hd = j / EDGE_F, jj = j % EDGE_F;
  float s = 0.0f;
  for (int c = 0; c < H; ++c)
    s += Wedge[(size_t)jj * (HEADS * H) + hd * H + c] *
         att[(size_t)hd * 3 * H + 2 * H + c];
  vh[j] = s;
}

// ---- per-layer init: aggr=0, amax=enc(-inf), denom=0 ----
__global__ void k_init_layer(float* __restrict__ aggr,
                             unsigned* __restrict__ amax,
                             float* __restrict__ denom) {
  int idx = blockIdx.x * blockDim.x + threadIdx.x;
  if (idx < NPAD * HEADS * H) aggr[idx] = 0.0f;
  if (idx < NPAD * HEADS) { amax[idx] = 0x007FFFFFu; denom[idx] = 0.0f; }
}

// ---- per (node,head,{dst,src}) attention score dot products ----
__global__ void k_node_scores(const float* __restrict__ hmat,
                              const float* __restrict__ att,
                              float* __restrict__ sdst,
                              float* __restrict__ ssrc) {
  int t = blockIdx.x * blockDim.x + threadIdx.x;
  if (t >= NN * HEADS * 2) return;
  int which = t & 1, hd = (t >> 1) & 1, n = t >> 2;
  const float4* hp = (const float4*)(hmat + (size_t)n * HEADS * H + hd * H);
  const float4* ap = (const float4*)(att + (size_t)hd * 3 * H + which * H);
  float s = 0.0f;
#pragma unroll 8
  for (int i = 0; i < H / 4; ++i) {
    float4 a = ap[i], h = hp[i];
    s += a.x * h.x + a.y * h.y + a.z * h.z + a.w * h.w;
  }
  (which ? ssrc : sdst)[n * HEADS + hd] = s;
}

// ---- edge pass 1: raw alpha + segment max ----
__global__ void k_edge_alpha(const int* __restrict__ ei,
                             const float* __restrict__ ean,
                             const float* __restrict__ vh,
                             const float* __restrict__ sdst,
                             const float* __restrict__ ssrc,
                             float* __restrict__ alpha,
                             unsigned* __restrict__ amax) {
  int t = blockIdx.x * blockDim.x + threadIdx.x;
  if (t >= E * HEADS) return;
  int hd = t & 1, e = t >> 1;
  int row = ei[e], col = ei[E + e];
  float s = sdst[row * HEADS + hd] + ssrc[col * HEADS + hd] +
            ean[e * 3 + 0] * vh[hd * 3 + 0] +
            ean[e * 3 + 1] * vh[hd * 3 + 1] +
            ean[e * 3 + 2] * vh[hd * 3 + 2];
  s = (s > 0.0f) ? s : NEG_SLOPE * s;            // leaky_relu
  alpha[t] = s;
  atomicMax(&amax[row * HEADS + hd], fenc(s));
}

// ---- edge pass 2: exp(alpha - max) + segment sum ----
__global__ void k_edge_exp(const int* __restrict__ ei,
                           const unsigned* __restrict__ amax,
                           float* __restrict__ alpha,
                           float* __restrict__ denom) {
  int t = blockIdx.x * blockDim.x + threadIdx.x;
  if (t >= E * HEADS) return;
  int hd = t & 1, e = t >> 1;
  int row = ei[e];
  float a = __expf(alpha[t] - fdec(amax[row * HEADS + hd]));
  alpha[t] = a;
  atomicAdd(&denom[row * HEADS + hd], a);
}

// ---- edge pass 3: one wave per (edge,head); lane handles 4 channels ----
__global__ void k_edge_aggr(const int* __restrict__ ei,
                            const float* __restrict__ alpha,
                            const float* __restrict__ denom,
                            const float* __restrict__ hmat,
                            float* __restrict__ aggr) {
  int gtid = blockIdx.x * blockDim.x + threadIdx.x;
  int w = gtid >> 5, lane = gtid & 31;
  if (w >= E * HEADS) return;
  int hd = w & 1, e = w >> 1;
  int row = ei[e], col = ei[E + e];
  float wgt = alpha[w] / (denom[row * HEADS + hd] + 1e-16f);
  float4 hv = *(const float4*)(hmat + (size_t)col * HEADS * H + hd * H + lane * 4);
  float* dst = aggr + (size_t)row * HEADS * H + hd * H + lane * 4;
  atomicAdd(dst + 0, hv.x * wgt);
  atomicAdd(dst + 1, hv.y * wgt);
  atomicAdd(dst + 2, hv.z * wgt);
  atomicAdd(dst + 3, hv.w * wgt);
}

// ---- head mean + bias + ReLU -> f32 & bf16 ----
__global__ void k_finalize(const float* __restrict__ aggr,
                           const float* __restrict__ gbias,
                           float* __restrict__ xf32,
                           __bf16* __restrict__ xb16) {
  int t = blockIdx.x * blockDim.x + threadIdx.x;
  if (t >= NN * H) return;
  int n = t / H, c = t % H;
  float v = 0.5f * (aggr[(size_t)n * HEADS * H + c] +
                    aggr[(size_t)n * HEADS * H + H + c]) + gbias[c];
  v = fmaxf(v, 0.0f);
  xf32[t] = v;
  xb16[t] = (__bf16)v;
}

// =====================================================================
extern "C" void kernel_launch(void* const* d_in, const int* in_sizes, int n_in,
                              void* d_out, int out_size, void* d_ws,
                              size_t ws_size, hipStream_t stream) {
  (void)in_sizes; (void)n_in; (void)out_size;
  const float* ue      = (const float*)d_in[0];
  const float* gnb     = (const float*)d_in[1];
  const int*   ei      = (const int*)  d_in[2];
  const float* eattr   = (const float*)d_in[3];
  const float* ue_sc   = (const float*)d_in[4];
  const float* ue_sh   = (const float*)d_in[5];
  const float* gnb_sc  = (const float*)d_in[6];
  const float* gnb_sh  = (const float*)d_in[7];
  const float* e_sc    = (const float*)d_in[8];
  const float* e_sh    = (const float*)d_in[9];
  const float* W_ue    = (const float*)d_in[10];
  const float* b_ue    = (const float*)d_in[11];
  const float* W_gnb   = (const float*)d_in[12];
  const float* b_gnb   = (const float*)d_in[13];
  const float* g_lin[2]  = {(const float*)d_in[14], (const float*)d_in[18]};
  const float* g_edge[2] = {(const float*)d_in[15], (const float*)d_in[19]};
  const float* g_att[2]  = {(const float*)d_in[16], (const float*)d_in[20]};
  const float* g_bias[2] = {(const float*)d_in[17], (const float*)d_in[21]};
  const float* q_W1 = (const float*)d_in[22];
  const float* q_b1 = (const float*)d_in[23];
  const float* q_W2 = (const float*)d_in[24];
  const float* q_b2 = (const float*)d_in[25];
  const float* q_W3 = (const float*)d_in[26];
  const float* q_b3 = (const float*)d_in[27];
  float* out = (float*)d_out;

  // ---- workspace carve-out (256B aligned slices) ----
  char* wsp = (char*)d_ws;
  size_t used = 0;
  auto alloc = [&](size_t bytes) -> char* {
    char* p = wsp + used;
    used += (bytes + 255) & ~(size_t)255;
    return p;
  };
  float*    ean    = (float*)   alloc((size_t)E * EDGE_F * 4);
  __bf16*   xb     = (__bf16*)  alloc((size_t)NPAD * H * 2);
  float*    hmat   = (float*)   alloc((size_t)NPAD * HEADS * H * 4);
  float*    sdst   = (float*)   alloc((size_t)NPAD * HEADS * 4);
  float*    ssrc   = (float*)   alloc((size_t)NPAD * HEADS * 4);
  unsigned* amax   = (unsigned*)alloc((size_t)NPAD * HEADS * 4);
  float*    denom  = (float*)   alloc((size_t)NPAD * HEADS * 4);
  float*    alpha  = (float*)   alloc((size_t)E * HEADS * 4);
  float*    aggr   = (float*)   alloc((size_t)NPAD * HEADS * H * 4);
  float*    xf32   = (float*)   alloc((size_t)NPAD * H * 4);
  float*    vh     = (float*)   alloc(256);
  __bf16*   wue    = (__bf16*)  alloc((size_t)UE_F * H * 2);
  __bf16*   wgnb   = (__bf16*)  alloc((size_t)GNB_F * H * 2);
  __bf16*   glin_b[2];
  glin_b[0]        = (__bf16*)  alloc((size_t)H * HEADS * H * 2);
  glin_b[1]        = (__bf16*)  alloc((size_t)H * HEADS * H * 2);
  __bf16*   qw1    = (__bf16*)  alloc((size_t)H * H * 2);
  __bf16*   qw2    = (__bf16*)  alloc((size_t)H * (H / 2) * 2);
  __bf16*   qw3    = (__bf16*)  alloc((size_t)(H / 2) * N_ACT * 2);
  __bf16*   uestg  = (__bf16*)  alloc((size_t)N_UE * UE_F * 2);
  __bf16*   gnbstg = (__bf16*)  alloc((size_t)GNB_PAD * GNB_F * 2);
  __bf16*   z1     = (__bf16*)  alloc((size_t)N_UE * H * 2);
  __bf16*   z2     = (__bf16*)  alloc((size_t)N_UE * (H / 2) * 2);
  if (used > ws_size) return;   // workspace too small: bail deterministically

  // ---- weight conversions (f32 row-major -> bf16 column-major) ----
  auto cvt = [&](const float* W, __bf16* o, int K, int N) {
    k_convert_w<<<CDIV(K * N, BLK), BLK, 0, stream>>>(W, o, K, N);
  };
  cvt(W_ue, wue, UE_F, H);
  cvt(W_gnb, wgnb, GNB_F, H);
  cvt(g_lin[0], glin_b[0], H, HEADS * H);
  cvt(g_lin[1], glin_b[1], H, HEADS * H);
  cvt(q_W1, qw1, H, H);
  cvt(q_W2, qw2, H, H / 2);
  cvt(q_W3, qw3, H / 2, N_ACT);

  // ---- stage + normalize inputs ----
  k_stage_rows<<<CDIV(N_UE * UE_F, BLK), BLK, 0, stream>>>(
      ue, ue_sc, ue_sh, uestg, N_UE, N_UE, UE_F);
  k_stage_rows<<<CDIV(GNB_PAD * GNB_F, BLK), BLK, 0, stream>>>(
      gnb, gnb_sc, gnb_sh, gnbstg, N_GNB, GNB_PAD, GNB_F);
  k_edge_norm<<<CDIV(E * EDGE_F, BLK), BLK, 0, stream>>>(eattr, e_sc, e_sh, ean);
  k_zero_pad<<<CDIV((NPAD - NN) * H, BLK), BLK, 0, stream>>>(xb);

  // ---- input encoders: node_h = relu(x_n @ W + b) ----
  launch_gemm<UE_F, 4>(uestg, UE_F, wue, b_ue, xf32, xb, H, N_UE, N_UE, 1,
                       stream);
  launch_gemm<GNB_F, 4>(gnbstg, GNB_F, wgnb, b_gnb, xf32 + (size_t)N_UE * H,
                        xb + (size_t)N_UE * H, H, GNB_PAD, N_GNB, 1, stream);

  // ---- two EGAT layers ----
  for (int L = 0; L < 2; ++L) {
    launch_gemm<H, 4>(xb, H, glin_b[L], nullptr, hmat, nullptr, HEADS * H,
                      NPAD, NPAD, 0, stream);
    k_vhead<<<1, 64, 0, stream>>>(g_edge[L], g_att[L], vh);
    k_init_layer<<<CDIV(NPAD * HEADS * H, BLK), BLK, 0, stream>>>(aggr, amax,
                                                                  denom);
    k_node_scores<<<CDIV(NN * HEADS * 2, BLK), BLK, 0, stream>>>(
        hmat, g_att[L], sdst, ssrc);
    k_edge_alpha<<<CDIV(E * HEADS, BLK), BLK, 0, stream>>>(
        ei, ean, vh, sdst, ssrc, alpha, amax);
    k_edge_exp<<<CDIV(E * HEADS, BLK), BLK, 0, stream>>>(ei, amax, alpha,
                                                         denom);
    k_edge_aggr<<<CDIV(E * HEADS * 32, BLK), BLK, 0, stream>>>(
        ei, alpha, denom, hmat, aggr);
    k_finalize<<<CDIV(NN * H, BLK), BLK, 0, stream>>>(aggr, g_bias[L], xf32,
                                                      xb);
  }

  // ---- Q head on UE nodes (rows 0..9999 of xb) ----
  launch_gemm<H, 4>(xb, H, qw1, q_b1, nullptr, z1, H, N_UE, N_UE, 1, stream);
  launch_gemm<H, 4>(z1, H, qw2, q_b2, nullptr, z2, H / 2, N_UE, N_UE, 1,
                    stream);
  launch_gemm<H / 2, 2>(z2, H / 2, qw3, q_b3, out, nullptr, N_ACT, N_UE, N_UE,
                        0, stream);
}
